// GNNEncoder_77068893159613
// MI455X (gfx1250) — matrix operand order repaired and checked by reading, not compiled, not run on hardware
//
#include <hip/hip_runtime.h>
#include <hip/hip_bf16.h>

// ---------------------------------------------------------------------------
// Types for CDNA5 WMMA (wave32): v_wmma_f32_16x16x32_bf16
// ---------------------------------------------------------------------------
typedef __attribute__((ext_vector_type(16))) __bf16 v16bf;
typedef __attribute__((ext_vector_type(8)))  __bf16 v8bf;
typedef __attribute__((ext_vector_type(8)))  float  v8f;

__device__ __forceinline__ __bf16 f2bf(float f) {
    unsigned u = __builtin_bit_cast(unsigned, f);
    // round-to-nearest-even truncation to bf16
    unsigned r = u + 0x7FFFu + ((u >> 16) & 1u);
    unsigned short h = (unsigned short)(r >> 16);
    return __builtin_bit_cast(__bf16, h);
}

// Guarantee the native fp32 atomic (GLOBAL_ATOMIC_ADD_F32), no CAS fallback.
__device__ __forceinline__ void atomic_add_f32(float* addr, float val) {
    asm volatile("global_atomic_add_f32 %0, %1, off"
                 :
                 : "v"(addr), "v"(val)
                 : "memory");
}

// ---------------------------------------------------------------------------
// Utility kernels
// ---------------------------------------------------------------------------
__global__ void fill_f32(float* __restrict__ p, float v, int n) {
    int i = blockIdx.x * blockDim.x + threadIdx.x;
    if (i < n) p[i] = v;
}

// deg[dst[e]] += 1  (deg pre-filled with 1.0 for the self loop)
__global__ void degree_kernel(const int* __restrict__ dst,
                              float* __restrict__ deg, int nE) {
    int e = blockIdx.x * blockDim.x + threadIdx.x;
    if (e < nE) atomic_add_f32(&deg[dst[e]], 1.0f);
}

__global__ void dinv_kernel(const float* __restrict__ deg,
                            float* __restrict__ dinv, int n) {
    int i = blockIdx.x * blockDim.x + threadIdx.x;
    if (i < n) dinv[i] = rsqrtf(fmaxf(deg[i], 1.0f));
}

// W[128,128] row-major f32 -> Wt[col*128 + k] bf16 (transposed, column-major)
__global__ void cvt_w_kernel(const float* __restrict__ W,
                             __bf16* __restrict__ Wt) {
    int idx = blockIdx.x * blockDim.x + threadIdx.x;  // idx = k*128 + col
    if (idx < 128 * 128) {
        int k = idx >> 7, col = idx & 127;
        Wt[col * 128 + k] = f2bf(W[idx]);
    }
}

__global__ void cvt_x_kernel(const float* __restrict__ X,
                             __bf16* __restrict__ Xb, int n) {
    int i = blockIdx.x * blockDim.x + threadIdx.x;
    if (i < n) Xb[i] = f2bf(X[i]);
}

// ---------------------------------------------------------------------------
// WMMA GEMM: Out[N,128] = Xb[N,128](bf16) @ W(bf16, stored transposed
// Wt[col*128+k]).  One wave32 computes one 16x16 output tile; K-loop of 4
// v_wmma_f32_16x16x32_bf16 with f32 accumulation.
// Grid: (N/16, 2), block 128 (4 waves); colTile = blockIdx.y*4 + wave (0..7).
//
// VGPR layouts used here (C/D as in CDNA5 ISA 7.12.2):
//   A 16x32 bf16: lane&15 = row, lane>>4 selects K-half of 16 (K = k0+16*(lane>>4)+s);
//                 contiguous K order, same as B (not the ISA table order).
//   B 32x16 bf16: lane&15 = column, lane>>4 selects K-half of 16 (contiguous, matching A).
//   C/D f32:      VGPR r -> row r (+8 for lanes 16..31), lane&15 = column.
// ---------------------------------------------------------------------------
__global__ void gemm_bf16_wmma(const __bf16* __restrict__ Xb,
                               const __bf16* __restrict__ Wt,
                               float* __restrict__ Out) {
    const int lane  = threadIdx.x & 31;
    const int wave  = threadIdx.x >> 5;
    const int lrow  = lane & 15;
    const int khalf = lane >> 4;

    const int rowBase = blockIdx.x * 16;
    const int colBase = (blockIdx.y * 4 + wave) * 16;

    const __bf16* xr = Xb + (rowBase + lrow) * 128;
    const __bf16* wc = Wt + (colBase + lrow) * 128;

    v8f acc = {};
#pragma unroll
    for (int k0 = 0; k0 < 128; k0 += 32) {
        // A operand: one contiguous 16-element (32B) load per lane
        v16bf a = *(const v16bf*)(xr + k0 + khalf * 16);

        // B operand: one contiguous 16-element (32B) load per lane
        v16bf b = *(const v16bf*)(wc + k0 + khalf * 16);

        acc = __builtin_amdgcn_wmma_f32_16x16x32_bf16(
            /*neg_a=*/false, a, /*neg_b=*/false, b,
            /*c_mod=*/(short)0, acc, /*reuse_a=*/false, /*reuse_b=*/false);
    }

    float* orow = Out + (rowBase + khalf * 8) * 128 + colBase + lrow;
#pragma unroll
    for (int r = 0; r < 8; ++r) orow[r * 128] = acc[r];
}

// ---------------------------------------------------------------------------
// Edge scatter: one wave per (edge | self-loop); lane handles 4 channels.
// Agg[dst] += T[src] * dinv[src]*dinv[dst]   via native fp32 atomics.
// ---------------------------------------------------------------------------
__global__ void scatter_kernel(const float* __restrict__ T,
                               const int* __restrict__ src,
                               const int* __restrict__ dst,
                               const float* __restrict__ dinv,
                               float* __restrict__ Agg,
                               int nE, int nN) {
    int tid  = blockIdx.x * blockDim.x + threadIdx.x;
    int e    = tid >> 5;
    int lane = tid & 31;
    if (e >= nE + nN) return;

    int s, d;
    if (e < nE) { s = src[e]; d = dst[e]; }
    else        { s = d = e - nE; }                 // self loop

    float norm = dinv[s] * dinv[d];
    float4 v = ((const float4*)(T + s * 128))[lane];  // B128 gather (L2 hit)
    float* o = Agg + d * 128 + lane * 4;
    atomic_add_f32(o + 0, v.x * norm);
    atomic_add_f32(o + 1, v.y * norm);
    atomic_add_f32(o + 2, v.z * norm);
    atomic_add_f32(o + 3, v.w * norm);
}

// h = PReLU(agg + b1) written directly as bf16 (A-operand of layer-2 GEMM)
__global__ void prelu_bias_bf16(const float* __restrict__ agg,
                                const float* __restrict__ b,
                                const float* __restrict__ a,
                                __bf16* __restrict__ Hb, int n) {
    int i = blockIdx.x * blockDim.x + threadIdx.x;
    if (i < n) {
        int c = i & 127;
        float v = agg[i] + b[c];
        v = v > 0.0f ? v : a[c] * v;
        Hb[i] = f2bf(v);
    }
}

__global__ void bias_add(float* __restrict__ out,
                         const float* __restrict__ b, int n) {
    int i = blockIdx.x * blockDim.x + threadIdx.x;
    if (i < n) out[i] += b[i & 127];
}

// ---------------------------------------------------------------------------
// Launch
// ---------------------------------------------------------------------------
extern "C" void kernel_launch(void* const* d_in, const int* in_sizes, int n_in,
                              void* d_out, int out_size, void* d_ws, size_t ws_size,
                              hipStream_t stream) {
    const float* x  = (const float*)d_in[0];
    const int*   ei = (const int*)  d_in[1];
    const float* W1 = (const float*)d_in[2];
    const float* b1 = (const float*)d_in[3];
    const float* W2 = (const float*)d_in[4];
    const float* b2 = (const float*)d_in[5];
    const float* pa = (const float*)d_in[6];

    const int N  = in_sizes[0] / 128;   // 10000
    const int E  = in_sizes[1] / 2;     // 640000
    const int NF = N * 128;
    const int* srcIdx = ei;
    const int* dstIdx = ei + E;
    float* out = (float*)d_out;

    // Workspace partition (floats / bf16), all offsets 16B-aligned for N=10000
    float*  deg  = (float*)d_ws;            // N
    float*  dinv = deg + N;                 // N
    __bf16* W1t  = (__bf16*)(dinv + N);     // 128*128
    __bf16* W2t  = W1t + 128 * 128;         // 128*128
    __bf16* Xbf  = W2t + 128 * 128;         // N*128 (reused as bf16 h)
    float*  bufA = (float*)(Xbf + NF);      // N*128 (t0, then t1)
    float*  bufB = bufA + NF;               // N*128 (agg1)

    const int TB = 256;
    dim3 gemmGrid(N / 16, 2);
    dim3 gemmBlock(128);
    int scatterBlocks = (E + N + 7) / 8;    // 8 edges (waves) per 256-thread block

    // --- init ---
    fill_f32<<<(N  + TB - 1) / TB, TB, 0, stream>>>(deg,  1.0f, N);   // self loop
    fill_f32<<<(NF + TB - 1) / TB, TB, 0, stream>>>(bufB, 0.0f, NF);
    fill_f32<<<(NF + TB - 1) / TB, TB, 0, stream>>>(out,  0.0f, NF);

    // --- degree + norm ---
    degree_kernel<<<(E + TB - 1) / TB, TB, 0, stream>>>(dstIdx, deg, E);
    dinv_kernel<<<(N + TB - 1) / TB, TB, 0, stream>>>(deg, dinv, N);

    // --- weight / input conversion ---
    cvt_w_kernel<<<(128 * 128 + TB - 1) / TB, TB, 0, stream>>>(W1, W1t);
    cvt_w_kernel<<<(128 * 128 + TB - 1) / TB, TB, 0, stream>>>(W2, W2t);
    cvt_x_kernel<<<(NF + TB - 1) / TB, TB, 0, stream>>>(x, Xbf, NF);

    // --- layer 1 ---
    gemm_bf16_wmma<<<gemmGrid, gemmBlock, 0, stream>>>(Xbf, W1t, bufA);
    scatter_kernel<<<scatterBlocks, TB, 0, stream>>>(bufA, srcIdx, dstIdx,
                                                     dinv, bufB, E, N);
    prelu_bias_bf16<<<(NF + TB - 1) / TB, TB, 0, stream>>>(bufB, b1, pa, Xbf, NF);

    // --- layer 2 ---
    gemm_bf16_wmma<<<gemmGrid, gemmBlock, 0, stream>>>(Xbf, W2t, bufA);
    scatter_kernel<<<scatterBlocks, TB, 0, stream>>>(bufA, srcIdx, dstIdx,
                                                     dinv, out, E, N);
    bias_add<<<(NF + TB - 1) / TB, TB, 0, stream>>>(out, b2, NF);
}
